// Static_Deep_GATE_Conv_11888469476247
// MI455X (gfx1250) — compile-verified, matrix operand order checked by reading, not run on hardware
//
#include <hip/hip_runtime.h>
#include <hip/hip_bf16.h>

#define N_NODES 50000
#define N_EDGES 1600000
#define NODE_IN 128
#define C_DIM   64
#define EDGE_IN 64
#define EO_DIM  32
#define H_DIM   2
#define ATT_DIM 64
#define HC      (H_DIM * C_DIM)     // 128
#define HEO     (H_DIM * EO_DIM)    // 64
#define FEAT    (2 * C_DIM + EO_DIM) // 160
#define NEG_SLOPE 0.2f

typedef __attribute__((ext_vector_type(16))) __bf16 v16bf;
typedef __attribute__((ext_vector_type(8)))  float  v8f;

static __device__ inline v8f wmma_bf16(v16bf a, v16bf b, v8f c) {
  // D = A(16x32 bf16) * B(32x16 bf16) + C(16x16 f32)
  return __builtin_amdgcn_wmma_f32_16x16x32_bf16(false, a, false, b,
                                                 (short)0, c, false, false);
}

// Load 16 contiguous f32 and convert to bf16 (A-operand fragment for one lane).
static __device__ inline v16bf load_cvt16(const float* p) {
  v16bf r;
#pragma unroll
  for (int i = 0; i < 16; ++i) r[i] = (__bf16)p[i];
  return r;
}

static __device__ inline void atomicMaxF(float* a, float v) {
  int old = __float_as_int(*a);
  while (__int_as_float(old) < v) {
    int assumed = old;
    old = atomicCAS((int*)a, assumed, __float_as_int(v));
    if (old == assumed) break;
  }
}

// ---------------- weight prep: bf16 transposes + fused edge-out weight -------
__global__ void prep_kernel(const float* Wl, const float* W_att, const float* We,
                            const float* W_nout, const float* W_eout,
                            const float* be, const float* b_eout,
                            __bf16* Wl_t, __bf16* Watt_t, __bf16* We_t,
                            __bf16* Wn_t, __bf16* Wef_t, float* b_ef) {
  const int tid = threadIdx.x;
  for (int i = tid; i < NODE_IN * HC; i += blockDim.x) {
    int k = i / HC, n = i % HC;
    Wl_t[n * NODE_IN + k] = (__bf16)Wl[i];
  }
  for (int i = tid; i < FEAT * ATT_DIM; i += blockDim.x) {
    int f = i / ATT_DIM, n = i % ATT_DIM;
    Watt_t[n * FEAT + f] = (__bf16)W_att[i];
  }
  for (int i = tid; i < EDGE_IN * HEO; i += blockDim.x) {
    int k = i / HEO, n = i % HEO;
    We_t[n * EDGE_IN + k] = (__bf16)We[i];
  }
  for (int i = tid; i < HC * C_DIM; i += blockDim.x) {
    int k = i / C_DIM, n = i % C_DIM;
    Wn_t[n * HC + k] = (__bf16)W_nout[i];
  }
  // fused: edge_out = ea @ (We @ W_eout) + (be @ W_eout + b_eout)
  for (int i = tid; i < EO_DIM * EDGE_IN; i += blockDim.x) {
    int n = i / EDGE_IN, k = i % EDGE_IN;
    float acc = 0.f;
    for (int j = 0; j < HEO; ++j) acc += We[k * HEO + j] * W_eout[j * EO_DIM + n];
    Wef_t[n * EDGE_IN + k] = (__bf16)acc;
  }
  for (int n = tid; n < EO_DIM; n += blockDim.x) {
    float acc = b_eout[n];
    for (int j = 0; j < HEO; ++j) acc += be[j] * W_eout[j * EO_DIM + n];
    b_ef[n] = acc;
  }
}

// ---------------- accumulator init ------------------------------------------
__global__ void init_kernel(float* amax, float* denom, float* nacc) {
  long i = (long)blockIdx.x * blockDim.x + threadIdx.x;
  const long stride = (long)gridDim.x * blockDim.x;
  for (long j = i; j < (long)N_NODES * HC; j += stride) nacc[j] = 0.f;
  for (long j = i; j < (long)N_NODES * H_DIM; j += stride) {
    amax[j]  = -__builtin_inff();
    denom[j] = 0.f;
  }
}

// ---------------- xl = x @ Wl + bl  (store bf16) -----------------------------
__global__ void __launch_bounds__(256)
xl_gemm_kernel(const float* x, const __bf16* Wl_t, const float* bl, __bf16* xl) {
  const int lane = threadIdx.x & 31, half = lane >> 4, l16 = lane & 15;
  const int rt = blockIdx.x;        // 16-row tile of nodes
  const int ct = threadIdx.y;       // 16-col tile (8 tiles = 128 outputs)
  const float*  arow = x    + (size_t)(rt * 16 + l16) * NODE_IN + half * 16;
  const __bf16* brow = Wl_t + (size_t)(ct * 16 + l16) * NODE_IN + half * 16;
  v8f acc = {0.f, 0.f, 0.f, 0.f, 0.f, 0.f, 0.f, 0.f};
#pragma unroll
  for (int k0 = 0; k0 < NODE_IN; k0 += 32) {
    v16bf a = load_cvt16(arow + k0);
    v16bf b = *(const v16bf*)(brow + k0);
    acc = wmma_bf16(a, b, acc);
  }
  const int n = ct * 16 + l16;
  const float bias = bl[n];
#pragma unroll
  for (int r = 0; r < 8; ++r) {
    const int m = rt * 16 + r + 8 * half;
    xl[(size_t)m * HC + n] = (__bf16)(acc[r] + bias);
  }
}

// ---------------- attention logits per edge ----------------------------------
__global__ void __launch_bounds__(256)
attn_kernel(const float* edge_attr, const int* edge_index, const __bf16* xl,
            const __bf16* We_t, const __bf16* Watt_t,
            const float* be, const float* b_att, const float* att,
            float* alpha, float* amax) {
  __shared__ __align__(32) __bf16 e_lds[8][16 * EO_DIM];  // per-wave e-tile
  const int lane = threadIdx.x & 31, half = lane >> 4, l16 = lane & 15;
  const int wv = threadIdx.y;
  const int t  = blockIdx.x * 8 + wv;     // 16-edge tile
  const int eidx = t * 16 + l16;          // this lane's A-row edge
  const int srcN = edge_index[eidx];
  const int dstN = edge_index[N_EDGES + eidx];
  const float*  ea_row = edge_attr + (size_t)eidx * EDGE_IN + half * 16;
  const __bf16* xi_row = xl + (size_t)dstN * HC;
  const __bf16* xj_row = xl + (size_t)srcN * HC;
  __bf16* elds = &e_lds[wv][0];

  const v8f vzero = {0.f, 0.f, 0.f, 0.f, 0.f, 0.f, 0.f, 0.f};

  for (int h = 0; h < H_DIM; ++h) {
    // --- e tile [16 x 32] = (edge_attr @ We + be)[:, h*32 : h*32+32] ---------
#pragma unroll
    for (int ct = 0; ct < 2; ++ct) {
      v8f acc = vzero;
#pragma unroll
      for (int k0 = 0; k0 < EDGE_IN; k0 += 32) {
        v16bf a = load_cvt16(ea_row + k0);
        v16bf b = *(const v16bf*)(We_t +
                   (size_t)(h * EO_DIM + ct * 16 + l16) * EDGE_IN + k0 + half * 16);
        acc = wmma_bf16(a, b, acc);
      }
      const float bias = be[h * EO_DIM + ct * 16 + l16];
#pragma unroll
      for (int r = 0; r < 8; ++r)
        elds[(r + 8 * half) * EO_DIM + ct * 16 + l16] = (__bf16)(acc[r] + bias);
    }

    // --- a = leaky_relu([x_i | x_j | e] @ W_att + b_att) ---------------------
    v8f accA[4] = {vzero, vzero, vzero, vzero};
#pragma unroll
    for (int chunk = 0; chunk < 5; ++chunk) {
      v16bf a;
      if (chunk < 2)
        a = *(const v16bf*)(xi_row + h * C_DIM + chunk * 32 + half * 16);
      else if (chunk < 4)
        a = *(const v16bf*)(xj_row + h * C_DIM + (chunk - 2) * 32 + half * 16);
      else
        a = *(const v16bf*)(elds + l16 * EO_DIM + half * 16);
#pragma unroll
      for (int nt = 0; nt < 4; ++nt) {
        v16bf b = *(const v16bf*)(Watt_t +
                   (size_t)(nt * 16 + l16) * FEAT + chunk * 32 + half * 16);
        accA[nt] = wmma_bf16(a, b, accA[nt]);
      }
    }

    // --- alpha = sum_n leaky_relu(a)*att[h,n] --------------------------------
    float contrib[8];
#pragma unroll
    for (int r = 0; r < 8; ++r) contrib[r] = 0.f;
#pragma unroll
    for (int nt = 0; nt < 4; ++nt) {
      const int n = nt * 16 + l16;
      const float bb = b_att[n];
      const float av = att[h * ATT_DIM + n];
#pragma unroll
      for (int r = 0; r < 8; ++r) {
        float aval = accA[nt][r] + bb;
        aval = aval > 0.f ? aval : aval * NEG_SLOPE;
        contrib[r] += aval * av;
      }
    }
    // reduce over the 16 lanes of each wave-half (wave32)
#pragma unroll
    for (int off = 1; off < 16; off <<= 1)
#pragma unroll
      for (int r = 0; r < 8; ++r)
        contrib[r] += __shfl_xor(contrib[r], off, 32);

    if (l16 == 0) {
#pragma unroll
      for (int r = 0; r < 8; ++r) {
        const int m  = r + 8 * half;
        const int eg = t * 16 + m;
        const float v = contrib[r];
        alpha[(size_t)eg * H_DIM + h] = v;
        const int d = edge_index[N_EDGES + eg];
        atomicMaxF(&amax[d * H_DIM + h], v);
      }
    }
  }
}

// ---------------- segment softmax: exp + denominator -------------------------
__global__ void softmax_ex_kernel(const int* edge_index, const float* amax,
                                  float* alpha, float* denom) {
  long i = (long)blockIdx.x * blockDim.x + threadIdx.x;
  const long stride = (long)gridDim.x * blockDim.x;
  for (; i < (long)N_EDGES * H_DIM; i += stride) {
    const long e = i >> 1;
    const int  h = (int)(i & 1);
    const int  d = edge_index[N_EDGES + e];
    float mx = amax[d * H_DIM + h];
    if (!(mx > -1e30f)) mx = 0.f;   // isfinite() guard (amax only -inf or finite)
    const float ex = __expf(alpha[i] - mx);
    alpha[i] = ex;
    atomicAdd(&denom[d * H_DIM + h], ex);
  }
}

// ---------------- scatter messages: out[dst] += x_j * alpha ------------------
__global__ void scatter_kernel(const int* edge_index, const float* alpha,
                               const float* denom, const __bf16* xl,
                               float* nacc) {
  long i = (long)blockIdx.x * blockDim.x + threadIdx.x;
  const long stride = (long)gridDim.x * blockDim.x;
  for (; i < (long)N_EDGES * C_DIM; i += stride) {
    const long e = i >> 6;
    const int  c = (int)(i & 63);
    const int  s = edge_index[e];
    const int  d = edge_index[N_EDGES + e];
#pragma unroll
    for (int h = 0; h < H_DIM; ++h) {
      const float w  = alpha[e * H_DIM + h] / (denom[d * H_DIM + h] + 1e-16f);
      const float xv = (float)xl[(size_t)s * HC + h * C_DIM + c];
      atomicAdd(&nacc[(size_t)d * HC + h * C_DIM + c], xv * w);
    }
  }
}

// ---------------- node_out = nacc @ W_nout + b_nout --------------------------
__global__ void __launch_bounds__(128)
nout_gemm_kernel(const float* nacc, const __bf16* Wn_t, const float* b_nout,
                 float* out) {
  const int lane = threadIdx.x & 31, half = lane >> 4, l16 = lane & 15;
  const int rt = blockIdx.x;
  const int ct = threadIdx.y;   // 4 col tiles = 64 outputs
  const float*  arow = nacc + (size_t)(rt * 16 + l16) * HC + half * 16;
  const __bf16* brow = Wn_t + (size_t)(ct * 16 + l16) * HC + half * 16;
  v8f acc = {0.f, 0.f, 0.f, 0.f, 0.f, 0.f, 0.f, 0.f};
#pragma unroll
  for (int k0 = 0; k0 < HC; k0 += 32) {
    acc = wmma_bf16(load_cvt16(arow + k0), *(const v16bf*)(brow + k0), acc);
  }
  const int n = ct * 16 + l16;
  const float bias = b_nout[n];
#pragma unroll
  for (int r = 0; r < 8; ++r)
    out[(size_t)(rt * 16 + r + 8 * half) * C_DIM + n] = acc[r] + bias;
}

// ---------------- edge_out = edge_attr @ Wef + b_ef --------------------------
__global__ void __launch_bounds__(256)
eout_gemm_kernel(const float* edge_attr, const __bf16* Wef_t, const float* b_ef,
                 float* out) {
  const int lane = threadIdx.x & 31, half = lane >> 4, l16 = lane & 15;
  const int wc   = blockIdx.x * 8 + threadIdx.y;
  const int tile = wc >> 1;
  const int ct   = wc & 1;      // 2 col tiles = 32 outputs
  const float*  arow = edge_attr + (size_t)(tile * 16 + l16) * EDGE_IN + half * 16;
  const __bf16* brow = Wef_t     + (size_t)(ct * 16 + l16) * EDGE_IN + half * 16;
  v8f acc = {0.f, 0.f, 0.f, 0.f, 0.f, 0.f, 0.f, 0.f};
#pragma unroll
  for (int k0 = 0; k0 < EDGE_IN; k0 += 32) {
    acc = wmma_bf16(load_cvt16(arow + k0), *(const v16bf*)(brow + k0), acc);
  }
  const int n = ct * 16 + l16;
  const float bias = b_ef[n];
#pragma unroll
  for (int r = 0; r < 8; ++r)
    out[(size_t)(tile * 16 + r + 8 * half) * EO_DIM + n] = acc[r] + bias;
}

// ---------------- host-side launch -------------------------------------------
extern "C" void kernel_launch(void* const* d_in, const int* in_sizes, int n_in,
                              void* d_out, int out_size, void* d_ws, size_t ws_size,
                              hipStream_t stream) {
  (void)in_sizes; (void)n_in; (void)out_size; (void)ws_size;
  const float* x          = (const float*)d_in[0];
  const float* edge_attr  = (const float*)d_in[1];
  const int*   edge_index = (const int*)  d_in[2];
  const float* Wl     = (const float*)d_in[3];
  const float* bl     = (const float*)d_in[4];
  const float* We     = (const float*)d_in[5];
  const float* be     = (const float*)d_in[6];
  const float* W_att  = (const float*)d_in[7];
  const float* b_att  = (const float*)d_in[8];
  const float* att    = (const float*)d_in[9];
  const float* W_nout = (const float*)d_in[10];
  const float* b_nout = (const float*)d_in[11];
  const float* W_eout = (const float*)d_in[12];
  const float* b_eout = (const float*)d_in[13];

  char* p = (char*)d_ws;
  auto alloc = [&](size_t bytes) -> char* {
    char* r = p;
    p += (bytes + 255) & ~(size_t)255;
    return r;
  };
  __bf16* Wl_t   = (__bf16*)alloc((size_t)NODE_IN * HC * 2);
  __bf16* Watt_t = (__bf16*)alloc((size_t)FEAT * ATT_DIM * 2);
  __bf16* We_t   = (__bf16*)alloc((size_t)EDGE_IN * HEO * 2);
  __bf16* Wn_t   = (__bf16*)alloc((size_t)HC * C_DIM * 2);
  __bf16* Wef_t  = (__bf16*)alloc((size_t)EDGE_IN * EO_DIM * 2);
  float*  b_ef   = (float*) alloc((size_t)EO_DIM * 4);
  __bf16* xl     = (__bf16*)alloc((size_t)N_NODES * HC * 2);
  float*  alpha  = (float*) alloc((size_t)N_EDGES * H_DIM * 4);
  float*  amax   = (float*) alloc((size_t)N_NODES * H_DIM * 4);
  float*  denom  = (float*) alloc((size_t)N_NODES * H_DIM * 4);
  float*  nacc   = (float*) alloc((size_t)N_NODES * HC * 4);

  float* node_out = (float*)d_out;                       // [N, C]
  float* edge_out = node_out + (size_t)N_NODES * C_DIM;  // [E, EO]

  prep_kernel<<<1, 256, 0, stream>>>(Wl, W_att, We, W_nout, W_eout, be, b_eout,
                                     Wl_t, Watt_t, We_t, Wn_t, Wef_t, b_ef);
  init_kernel<<<4096, 256, 0, stream>>>(amax, denom, nacc);
  xl_gemm_kernel<<<N_NODES / 16, dim3(32, 8), 0, stream>>>(x, Wl_t, bl, xl);
  attn_kernel<<<N_EDGES / (16 * 8), dim3(32, 8), 0, stream>>>(
      edge_attr, edge_index, xl, We_t, Watt_t, be, b_att, att, alpha, amax);
  softmax_ex_kernel<<<4096, 256, 0, stream>>>(edge_index, amax, alpha, denom);
  scatter_kernel<<<(int)(((long)N_EDGES * C_DIM + 255) / 256), 256, 0, stream>>>(
      edge_index, alpha, denom, xl, nacc);
  nout_gemm_kernel<<<N_NODES / 16, dim3(32, 4), 0, stream>>>(nacc, Wn_t, b_nout,
                                                             node_out);
  eout_gemm_kernel<<<(N_EDGES / 16) * 2 / 8, dim3(32, 8), 0, stream>>>(
      edge_attr, Wef_t, b_ef, edge_out);
}
